// Decoder_14697378087362
// MI455X (gfx1250) — compile-verified, matrix operand order checked by reading, not run on hardware
//
#include <hip/hip_runtime.h>
#include <hip/hip_bf16.h>

// ---------------------------------------------------------------------------
// CDNA5 (gfx1250) attention-LSTM decoder.
//  - All dense GEMMs use v_wmma_f32_16x16x32_f16 (wave32 WMMA).
//  - Weights are converted once per launch to f16 and pre-packed into the
//    per-lane B-fragment layout so each wave loads its fragment with two
//    contiguous 16-byte loads.
//  - Keys/values are converted to f16 and transposed to [B][C][S] so the
//    batched attention GEMVs stream coalesced from L2 (everything fits in
//    the 192MB L2, so per-step traffic never goes back to HBM).
// ---------------------------------------------------------------------------

typedef __attribute__((ext_vector_type(16))) _Float16 v16h;
typedef __attribute__((ext_vector_type(8)))  float    v8f;
typedef _Float16 half_t;

#define B_    64
#define S_    2048
#define T_    400
#define NC_   128
#define NH_   512
#define NINP_ 34

union HV { v16h v; float4 f4[2]; };

// A fragment (16x32 f16, M x K): lane L<16 holds row M=L, Ks {0..7,16..23};
// lane>=16 holds row L-16, Ks {8..15,24..31}. Two 16B loads per lane.
__device__ __forceinline__ v16h load_a_frag(const half_t* A, int lda, int row0,
                                            int k0, int lane) {
  const half_t* p = A + (size_t)(row0 + (lane & 15)) * lda + k0 + ((lane >> 4) << 3);
  HV t;
  t.f4[0] = *(const float4*)(p);
  t.f4[1] = *(const float4*)(p + 16);
  return t.v;
}

// B fragment: pre-packed, 512 halves per fragment, lane-major (16 halves/lane).
__device__ __forceinline__ v16h load_b_frag(const half_t* pk, size_t frag, int lane) {
  const half_t* p = pk + (frag << 9) + ((size_t)lane << 4);
  HV t;
  t.f4[0] = ((const float4*)p)[0];
  t.f4[1] = ((const float4*)p)[1];
  return t.v;
}

__device__ __forceinline__ v8f wmma16(v16h a, v16h b, v8f c) {
  return __builtin_amdgcn_wmma_f32_16x16x32_f16(false, a, false, b, (short)0, c,
                                                false, false);
}

__device__ __forceinline__ float sigf(float x) { return 1.0f / (1.0f + __expf(-x)); }

__device__ __forceinline__ float hash_u01(unsigned x) {
  x ^= x >> 17; x *= 0xed5ad4bbu;
  x ^= x >> 11; x *= 0xac4c1b51u;
  x ^= x >> 15; x *= 0x31848babu;
  x ^= x >> 14;
  return ((x >> 8) + 0.5f) * (1.0f / 16777216.0f);
}

// --------------------------- one-time prep kernels -------------------------

// Pack row-major f32 W[N][K] into WMMA B fragments (W^T tiles), f16.
__global__ void k_pack(const float* __restrict__ W, half_t* __restrict__ out,
                       int N, int K) {
  int Kt = K >> 5;
  int total = (N >> 4) * Kt * 512;
  for (int idx = blockIdx.x * blockDim.x + threadIdx.x; idx < total;
       idx += gridDim.x * blockDim.x) {
    int r    = idx & 511;
    int frag = idx >> 9;
    int lane = r >> 4;
    int j    = r & 15;
    int nt   = frag / Kt;
    int kt   = frag - nt * Kt;
    int n    = (nt << 4) + (lane & 15);
    int k    = (kt << 5) + (j & 7) + ((j >> 3) << 4) + ((lane >> 4) << 3);
    out[idx] = (half_t)W[(size_t)n * K + k];
  }
}

__global__ void k_addbias(const float* __restrict__ a, const float* __restrict__ b,
                          float* __restrict__ out, int n) {
  int i = blockIdx.x * blockDim.x + threadIdx.x;
  if (i < n) out[i] = a[i] + b[i];
}

__global__ void k_cvt(const float* __restrict__ src, half_t* __restrict__ dst, int n) {
  int i = blockIdx.x * blockDim.x + threadIdx.x;
  if (i < n) dst[i] = (half_t)src[i];
}

// [B][S][C] f32 -> [B][C][S] f16
__global__ void k_transpose(const float* __restrict__ src, half_t* __restrict__ dst) {
  size_t total = (size_t)B_ * NC_ * S_;
  for (size_t idx = (size_t)blockIdx.x * blockDim.x + threadIdx.x; idx < total;
       idx += (size_t)gridDim.x * blockDim.x) {
    size_t b = idx / ((size_t)NC_ * S_);
    size_t r = idx - b * (size_t)NC_ * S_;
    size_t c = r / S_;
    size_t s = r - c * S_;
    dst[idx] = (half_t)src[(b * S_ + s) * NC_ + c];
  }
}

__global__ void k_init(const float* h00, const float* h01, const float* h02,
                       const float* c00, const float* c01, const float* c02,
                       half_t* __restrict__ h16, float* __restrict__ cst) {
  int idx = blockIdx.x * blockDim.x + threadIdx.x;
  if (idx >= 3 * B_ * NH_) return;
  int l   = idx / (B_ * NH_);
  int r   = idx - l * (B_ * NH_);
  int col = r & (NH_ - 1);
  const float* h0 = (l == 0) ? h00 : ((l == 1) ? h01 : h02);
  const float* c0 = (l == 0) ? c00 : ((l == 1) ? c01 : c02);
  h16[(size_t)l * 2 * B_ * NH_ + r] = (half_t)h0[col];  // parity 0
  cst[idx] = c0[col];
}

// --------------------------- per-step kernels ------------------------------

// x = [emb[tok] | ctx] f16, [64][256]
__global__ void k_xbuild(const int* __restrict__ inputs, int t,
                         const half_t* __restrict__ emb_h,
                         const half_t* __restrict__ ctx16,
                         half_t* __restrict__ x0) {
  int idx = blockIdx.x * blockDim.x + threadIdx.x;
  if (idx >= B_ * 2 * NC_) return;
  int b   = idx >> 8;
  int col = idx & 255;
  half_t v;
  if (col < NC_) {
    int tok = inputs[b * T_ + t];
    v = emb_h[tok * NC_ + col];
  } else {
    v = ctx16[b * NC_ + (col - NC_)];
  }
  x0[idx] = v;
}

// Fused LSTM cell: gates = [Ax|Ah] @ [Wih;Whh]^T + bias, then cell update.
// grid (32 ntiles, 4 mtiles), block 128 = 4 waves; wave g computes gate g.
__global__ void k_lstm(const half_t* __restrict__ Ax, int ldax, int ktx,
                       const half_t* __restrict__ Ah,
                       const half_t* __restrict__ pkIh,
                       const half_t* __restrict__ pkHh,
                       const float* __restrict__ bias,
                       float* __restrict__ c, half_t* __restrict__ hOut) {
  int nt   = blockIdx.x;            // 0..31  (hidden column tile)
  int mt   = blockIdx.y;            // 0..3   (batch row tile)
  int lane = threadIdx.x & 31;
  int g    = threadIdx.x >> 5;      // gate: 0=i 1=f 2=g 3=o
  int nG   = g * 32 + nt;           // global 16-col tile in the 2048 gate dim

  float bv = bias[nG * 16 + (lane & 15)];
  v8f acc;
#pragma unroll
  for (int v = 0; v < 8; ++v) acc[v] = bv;

  for (int kt = 0; kt < ktx; ++kt) {
    v16h a = load_a_frag(Ax, ldax, mt * 16, kt * 32, lane);
    v16h b = load_b_frag(pkIh, (size_t)nG * ktx + kt, lane);
    acc = wmma16(a, b, acc);
  }
  for (int kt = 0; kt < 16; ++kt) {
    v16h a = load_a_frag(Ah, NH_, mt * 16, kt * 32, lane);
    v16h b = load_b_frag(pkHh, (size_t)nG * 16 + kt, lane);
    acc = wmma16(a, b, acc);
  }

  __shared__ float gbuf[4][8][32];
#pragma unroll
  for (int v = 0; v < 8; ++v) gbuf[g][v][lane] = acc[v];
  __syncthreads();

  int tid = threadIdx.x;
#pragma unroll
  for (int e0 = 0; e0 < 2; ++e0) {
    int e   = tid + e0 * 128;       // element in 16x16 tile, (v,lane) order
    int v   = e >> 5;
    int ln  = e & 31;
    int row = mt * 16 + v + ((ln >> 4) << 3);
    int col = nt * 16 + (ln & 15);
    float iv = gbuf[0][v][ln], fv = gbuf[1][v][ln];
    float gv = gbuf[2][v][ln], ov = gbuf[3][v][ln];
    float cold = c[row * NH_ + col];
    float cn   = sigf(fv) * cold + sigf(iv) * tanhf(gv);
    float hn   = sigf(ov) * tanhf(cn);
    c[row * NH_ + col]    = cn;
    hOut[row * NH_ + col] = (half_t)hn;
  }
}

// q[64][128] = h3 @ wq^T + bq  (WMMA, grid (8,4), one wave/block)
__global__ void k_q(const half_t* __restrict__ h3, const half_t* __restrict__ pkwq,
                    const float* __restrict__ bq, float* __restrict__ q) {
  int nt = blockIdx.x, mt = blockIdx.y, lane = threadIdx.x;
  float bv = bq[nt * 16 + (lane & 15)];
  v8f acc;
#pragma unroll
  for (int v = 0; v < 8; ++v) acc[v] = bv;
  for (int kt = 0; kt < 16; ++kt)
    acc = wmma16(load_a_frag(h3, NH_, mt * 16, kt * 32, lane),
                 load_b_frag(pkwq, (size_t)nt * 16 + kt, lane), acc);
#pragma unroll
  for (int v = 0; v < 8; ++v) {
    int row = mt * 16 + v + ((lane >> 4) << 3);
    int col = nt * 16 + (lane & 15);
    q[row * NC_ + col] = acc[v];
  }
}

// masked softmax attention energies; one block per batch row.
__global__ void k_energy(const float* __restrict__ q, const half_t* __restrict__ keysT,
                         const int* __restrict__ lens, float* __restrict__ attOut,
                         half_t* __restrict__ att16) {
  int b   = blockIdx.x;
  int tid = threadIdx.x;            // 256
  __shared__ float qs[NC_];
  __shared__ float red[256];
  if (tid < NC_) qs[tid] = q[b * NC_ + tid];
  __syncthreads();

  float acc[8];
#pragma unroll
  for (int i = 0; i < 8; ++i) acc[i] = 0.f;
  const half_t* kp = keysT + (size_t)b * NC_ * S_;
  for (int c = 0; c < NC_; ++c) {
    float qc = qs[c];
    const half_t* kr = kp + (size_t)c * S_ + tid;
#pragma unroll
    for (int i = 0; i < 8; ++i) acc[i] += qc * (float)kr[i * 256];
  }
  int len = lens[b];
  float mx = -1e30f;
#pragma unroll
  for (int i = 0; i < 8; ++i) {
    int s = tid + i * 256;
    if (s >= len) acc[i] = -1e30f;
    mx = fmaxf(mx, acc[i]);
  }
  red[tid] = mx; __syncthreads();
  for (int o = 128; o > 0; o >>= 1) {
    if (tid < o) red[tid] = fmaxf(red[tid], red[tid + o]);
    __syncthreads();
  }
  mx = red[0]; __syncthreads();

  float ex[8], sum = 0.f;
#pragma unroll
  for (int i = 0; i < 8; ++i) {
    int s = tid + i * 256;
    ex[i] = (s < len) ? __expf(acc[i] - mx) : 0.f;
    sum += ex[i];
  }
  red[tid] = sum; __syncthreads();
  for (int o = 128; o > 0; o >>= 1) {
    if (tid < o) red[tid] += red[tid + o];
    __syncthreads();
  }
  float inv = 1.0f / red[0];
#pragma unroll
  for (int i = 0; i < 8; ++i) {
    int s = tid + i * 256;
    float a = ex[i] * inv;
    attOut[(size_t)b * S_ + s] = a;
    att16[b * S_ + s] = (half_t)a;
  }
}

// ctx[b][c] = sum_s att[b][s] * values[b][c][s]; one block per batch.
__global__ void k_ctx(const half_t* __restrict__ att16, const half_t* __restrict__ valsT,
                      half_t* __restrict__ ctx16) {
  int b = blockIdx.x;
  int c = threadIdx.x;              // 128
  __shared__ float lat[256];
  float acc = 0.f;
  const half_t* vp = valsT + ((size_t)b * NC_ + c) * S_;
  for (int s0 = 0; s0 < S_; s0 += 256) {
    lat[c]       = (float)att16[b * S_ + s0 + c];
    lat[c + 128] = (float)att16[b * S_ + s0 + c + 128];
    __syncthreads();
#pragma unroll 8
    for (int j = 0; j < 256; ++j) acc += lat[j] * (float)vp[s0 + j];
    __syncthreads();
  }
  ctx16[b * NC_ + c] = (half_t)acc;
}

// z = leaky_relu([h3|ctx] @ wc^T + bc), f16 out. grid (8,4), one wave/block.
__global__ void k_z(const half_t* __restrict__ h3, const half_t* __restrict__ ctx16,
                    const half_t* __restrict__ pkwc, const float* __restrict__ bc,
                    half_t* __restrict__ z16) {
  int nt = blockIdx.x, mt = blockIdx.y, lane = threadIdx.x;
  float bv = bc[nt * 16 + (lane & 15)];
  v8f acc;
#pragma unroll
  for (int v = 0; v < 8; ++v) acc[v] = bv;
  for (int kt = 0; kt < 16; ++kt)                       // h3 part, K=512
    acc = wmma16(load_a_frag(h3, NH_, mt * 16, kt * 32, lane),
                 load_b_frag(pkwc, (size_t)nt * 20 + kt, lane), acc);
  for (int kt = 16; kt < 20; ++kt)                      // ctx part, K=128
    acc = wmma16(load_a_frag(ctx16, NC_, mt * 16, (kt - 16) * 32, lane),
                 load_b_frag(pkwc, (size_t)nt * 20 + kt, lane), acc);
#pragma unroll
  for (int v = 0; v < 8; ++v) {
    float x = acc[v];
    x = (x > 0.f) ? x : 0.01f * x;
    int row = mt * 16 + v + ((lane >> 4) << 3);
    int col = nt * 16 + (lane & 15);
    z16[row * NC_ + col] = (half_t)x;
  }
}

// logits + gumbel argmax; one block per batch row.
__global__ void k_logits(const half_t* __restrict__ z16, const half_t* __restrict__ emb_h,
                         const float* __restrict__ b_out, float* __restrict__ logitsOut,
                         float* __restrict__ genOut, int t) {
  int b   = blockIdx.x;
  int tid = threadIdx.x;            // 64
  __shared__ float zrow[NC_];
  __shared__ float sc[64];
  __shared__ int   scj[64];
  zrow[tid]      = (float)z16[b * NC_ + tid];
  zrow[tid + 64] = (float)z16[b * NC_ + tid + 64];
  __syncthreads();
  float score = -1e30f;
  if (tid < NINP_) {
    float acc = b_out[tid];
    for (int c = 0; c < NC_; ++c) acc += zrow[c] * (float)emb_h[tid * NC_ + c];
    logitsOut[((size_t)t * B_ + b) * NINP_ + tid] = acc;
    float u  = hash_u01(((unsigned)t * B_ + b) * NINP_ + tid + 0x9e3779b9u);
    float gn = -logf(1e-10f - logf(u + 1e-10f));
    score = acc + gn;
  }
  sc[tid] = score; scj[tid] = tid; __syncthreads();
  for (int o = 32; o > 0; o >>= 1) {
    if (tid < o && sc[tid + o] > sc[tid]) { sc[tid] = sc[tid + o]; scj[tid] = scj[tid + o]; }
    __syncthreads();
  }
  if (tid == 0) genOut[(size_t)t * B_ + b] = (float)scj[0];
}

// --------------------------------- host ------------------------------------

extern "C" void kernel_launch(void* const* d_in, const int* in_sizes, int n_in,
                              void* d_out, int out_size, void* d_ws, size_t ws_size,
                              hipStream_t stream) {
  (void)in_sizes; (void)n_in; (void)out_size; (void)ws_size;

  const float* keys   = (const float*)d_in[0];
  const float* values = (const float*)d_in[1];
  const int*   lens   = (const int*)d_in[2];
  const int*   inputs = (const int*)d_in[3];
  const float* emb    = (const float*)d_in[4];
  const float* wq     = (const float*)d_in[5];
  const float* bq     = (const float*)d_in[6];
  const float* wc     = (const float*)d_in[7];
  const float* bc     = (const float*)d_in[8];
  const float* b_out  = (const float*)d_in[9];
  const float *w_ih[3], *w_hh[3], *b_ih[3], *b_hh[3], *h0[3], *c0[3];
  for (int l = 0; l < 3; ++l) {
    w_ih[l] = (const float*)d_in[10 + 6 * l + 0];
    w_hh[l] = (const float*)d_in[10 + 6 * l + 1];
    b_ih[l] = (const float*)d_in[10 + 6 * l + 2];
    b_hh[l] = (const float*)d_in[10 + 6 * l + 3];
    h0[l]   = (const float*)d_in[10 + 6 * l + 4];
    c0[l]   = (const float*)d_in[10 + 6 * l + 5];
  }

  char* w = (char*)d_ws;
  size_t off = 0;
  auto alloc = [&](size_t bytes) -> size_t {
    off = (off + 255) & ~(size_t)255;
    size_t o = off; off += bytes; return o;
  };
  const int KinL[3] = {2 * NC_, NH_, NH_};
  size_t o_pkih[3], o_pkhh[3];
  for (int l = 0; l < 3; ++l) {
    o_pkih[l] = alloc((size_t)4 * NH_ * KinL[l] * 2);
    o_pkhh[l] = alloc((size_t)4 * NH_ * NH_ * 2);
  }
  size_t o_pkwq  = alloc((size_t)NC_ * NH_ * 2);
  size_t o_pkwc  = alloc((size_t)NC_ * (NH_ + NC_) * 2);
  size_t o_embh  = alloc((size_t)NINP_ * NC_ * 2);
  size_t o_bias  = alloc((size_t)3 * 4 * NH_ * 4);
  size_t o_keysT = alloc((size_t)B_ * NC_ * S_ * 2);
  size_t o_valsT = alloc((size_t)B_ * NC_ * S_ * 2);
  size_t o_h16   = alloc((size_t)3 * 2 * B_ * NH_ * 2);
  size_t o_c     = alloc((size_t)3 * B_ * NH_ * 4);
  size_t o_x0    = alloc((size_t)B_ * 2 * NC_ * 2);
  size_t o_q     = alloc((size_t)B_ * NC_ * 4);
  size_t o_att16 = alloc((size_t)B_ * S_ * 2);
  size_t o_ctx16 = alloc((size_t)B_ * NC_ * 2);
  size_t o_z16   = alloc((size_t)B_ * NC_ * 2);
  size_t o_ascr  = alloc((size_t)B_ * S_ * 4);

  half_t* pkih[3]; half_t* pkhh[3];
  for (int l = 0; l < 3; ++l) {
    pkih[l] = (half_t*)(w + o_pkih[l]);
    pkhh[l] = (half_t*)(w + o_pkhh[l]);
  }
  half_t* pkwq  = (half_t*)(w + o_pkwq);
  half_t* pkwc  = (half_t*)(w + o_pkwc);
  half_t* embh  = (half_t*)(w + o_embh);
  float*  biasc = (float*)(w + o_bias);
  half_t* keysT = (half_t*)(w + o_keysT);
  half_t* valsT = (half_t*)(w + o_valsT);
  half_t* h16   = (half_t*)(w + o_h16);
  float*  cst   = (float*)(w + o_c);
  half_t* x0    = (half_t*)(w + o_x0);
  float*  qbuf  = (float*)(w + o_q);
  half_t* att16 = (half_t*)(w + o_att16);
  half_t* ctx16 = (half_t*)(w + o_ctx16);
  half_t* z16   = (half_t*)(w + o_z16);
  float*  ascr  = (float*)(w + o_ascr);

  auto h16p = [&](int l, int par) -> half_t* {
    return h16 + ((size_t)l * 2 + par) * B_ * NH_;
  };

  float* out      = (float*)d_out;
  float* o_logits = out;
  float* o_att    = out + (size_t)T_ * B_ * NINP_;
  float* o_gen    = o_att + (size_t)T_ * B_ * S_;

  // ---- one-time prep ----
  for (int l = 0; l < 3; ++l) {
    k_pack<<<1024, 256, 0, stream>>>(w_ih[l], pkih[l], 4 * NH_, KinL[l]);
    k_pack<<<1024, 256, 0, stream>>>(w_hh[l], pkhh[l], 4 * NH_, NH_);
    k_addbias<<<(4 * NH_ + 255) / 256, 256, 0, stream>>>(b_ih[l], b_hh[l],
                                                         biasc + l * 4 * NH_, 4 * NH_);
  }
  k_pack<<<256, 256, 0, stream>>>(wq, pkwq, NC_, NH_);
  k_pack<<<256, 256, 0, stream>>>(wc, pkwc, NC_, NH_ + NC_);
  k_cvt<<<(NINP_ * NC_ + 255) / 256, 256, 0, stream>>>(emb, embh, NINP_ * NC_);
  k_transpose<<<4096, 256, 0, stream>>>(keys, keysT);
  k_transpose<<<4096, 256, 0, stream>>>(values, valsT);
  k_init<<<(3 * B_ * NH_ + 255) / 256, 256, 0, stream>>>(h0[0], h0[1], h0[2],
                                                         c0[0], c0[1], c0[2], h16, cst);

  // ---- initial attention from h3 = h0 (att discarded -> scratch) ----
  k_q<<<dim3(8, 4), 32, 0, stream>>>(h16p(2, 0), pkwq, bq, qbuf);
  k_energy<<<B_, 256, 0, stream>>>(qbuf, keysT, lens, ascr, att16);
  k_ctx<<<B_, 128, 0, stream>>>(att16, valsT, ctx16);

  // ---- T sequential decode steps ----
  for (int t = 0; t < T_; ++t) {
    int p = t & 1;  // read parity for per-layer hidden state
    k_xbuild<<<(B_ * 2 * NC_ + 255) / 256, 256, 0, stream>>>(inputs, t, embh, ctx16, x0);
    k_lstm<<<dim3(32, 4), 128, 0, stream>>>(x0, 2 * NC_, 8, h16p(0, p),
                                            pkih[0], pkhh[0], biasc + 0,
                                            cst + 0, h16p(0, 1 - p));
    k_lstm<<<dim3(32, 4), 128, 0, stream>>>(h16p(0, 1 - p), NH_, 16, h16p(1, p),
                                            pkih[1], pkhh[1], biasc + 4 * NH_,
                                            cst + B_ * NH_, h16p(1, 1 - p));
    k_lstm<<<dim3(32, 4), 128, 0, stream>>>(h16p(1, 1 - p), NH_, 16, h16p(2, p),
                                            pkih[2], pkhh[2], biasc + 8 * NH_,
                                            cst + 2 * B_ * NH_, h16p(2, 1 - p));
    k_q<<<dim3(8, 4), 32, 0, stream>>>(h16p(2, 1 - p), pkwq, bq, qbuf);
    k_energy<<<B_, 256, 0, stream>>>(qbuf, keysT, lens,
                                     o_att + (size_t)t * B_ * S_, att16);
    k_ctx<<<B_, 128, 0, stream>>>(att16, valsT, ctx16);
    k_z<<<dim3(8, 4), 32, 0, stream>>>(h16p(2, 1 - p), ctx16, pkwc, bc, z16);
    k_logits<<<B_, 64, 0, stream>>>(z16, embh, b_out, o_logits, o_gen, t);
  }
}
